// CollaborativeWaterfallMoE_74105365725741
// MI455X (gfx1250) — matrix operand order repaired and checked by reading, not compiled, not run on hardware
//
#include <hip/hip_runtime.h>
#include <hip/hip_bf16.h>
#include <math.h>
#include <stdint.h>

typedef __attribute__((ext_vector_type(16))) __bf16 v16bf;
typedef __attribute__((ext_vector_type(8)))  __bf16 bf16x8;
typedef __attribute__((ext_vector_type(8)))  float  v8f;
typedef __attribute__((ext_vector_type(4)))  float  f32x4;

#define MT 64      // block tile M
#define NTB 64     // block tile N
#define KT 32      // K step (bf16 WMMA K)
#define LDA 40     // LDS row stride (halves): 80B, 16B-aligned, conflict-free
#define LDAF 36    // f32 LDS row stride (floats): 144B, 16B-aligned

// Build A fragment (16-bit 16x32 layout): halves 0..7 = K hiA..hiA+7,
// halves 8..15 = K hiA+16..hiA+23, hiA = (lane>=16)*8. K-contiguous -> 2x b128.
__device__ inline v16bf frag_a_bf16(const __bf16* row, int hiA)
{
    bf16x8 a0 = *(const bf16x8*)(row + hiA);
    bf16x8 a1 = *(const bf16x8*)(row + hiA + 16);
    v16bf af;
#pragma unroll
    for (int j = 0; j < 8; ++j) { af[j] = a0[j]; af[8 + j] = a1[j]; }
    return af;
}
// B fragment (32x16): halves 0..7 = K hiB..hiB+7, halves 8..15 = K hiB+16..hiB+23, hiB = (lane>=16)*8.
__device__ inline v16bf frag_b_bf16(const __bf16* row, int hiB)
{
    bf16x8 b0 = *(const bf16x8*)(row + hiB);
    bf16x8 b1 = *(const bf16x8*)(row + hiB + 16);
    v16bf bf;
#pragma unroll
    for (int j = 0; j < 8; ++j) { bf[j] = b0[j]; bf[8 + j] = b1[j]; }
    return bf;
}

// ---------------------------------------------------------------------------
// Implicit-GEMM 3x3 SAME conv, bf16 WMMA 16x16x32, f32 accumulate.
// 256 thr = 8 waves as 4(M) x 2(N-group); each wave: 16x32 (2 accumulators).
// Branch-free im2col gather: clamped unconditional loads + cndmask zeroing.
// POOL=1: fused ReLU + 8x8 mean pool (scorer, logW==5) via LDS partials.
// ---------------------------------------------------------------------------
template<int POOL>
__global__ __launch_bounds__(256)
void conv3x3_wmma(const float* __restrict__ X, const float* __restrict__ Wt,
                  const float* __restrict__ bias, float* __restrict__ Y,
                  float* __restrict__ pooled,
                  int Nimg, int Cin, int Cout, int logW,
                  float scale, int doRelu)
{
    __shared__ __align__(16) __bf16 lsA [MT][LDA];   // [m][k]
    __shared__ __align__(16) __bf16 lsBT[NTB][LDA];  // [n][k]
    __shared__ float spool[NTB][4];

    const int W     = 1 << logW;
    const int H     = W;
    const int logHW = 2 * logW;
    const int HW    = 1 << logHW;
    const int Ktot  = Cin * 9;
    const int mBase = blockIdx.x * MT;
    const int nBase = blockIdx.y * NTB;
    const int tid   = threadIdx.x;
    const int lane  = tid & 31;
    const int wv    = tid >> 5;
    const int wm    = (wv & 3) * 16;        // wave M offset in tile
    const int wng   = (wv >> 2) * 32;       // wave N-group offset in tile

    if (POOL) spool[tid >> 2][tid & 3] = 0.0f;

    v8f acc0 = {}, acc1 = {};

    for (int k0 = 0; k0 < Ktot; k0 += KT) {
        // ---- stage A: 64x32 im2col gather, m-fast for coalescing ----
        float va[8];
#pragma unroll
        for (int j = 0; j < 8; ++j) {
            int i  = tid + j * 256;
            int r  = i & 63;            // m within tile (lane-contiguous)
            int kk = i >> 6;            // k within tile
            int m  = mBase + r;
            int k  = k0 + kk;
            int img = m >> logHW;
            int rem = m & (HW - 1);
            int oy  = rem >> logW;
            int ox  = rem & (W - 1);
            int ci  = k / 9;
            int k9  = k - ci * 9;
            int dy  = k9 / 3 - 1;
            int dx  = k9 - (k9 / 3) * 3 - 1;
            int iy = oy + dy, ix = ox + dx;
            int ok = (k < Ktot) & (iy >= 0) & (iy < H) & (ix >= 0) & (ix < W);
            int iyc = min(max(iy, 0), H - 1);
            int ixc = min(max(ix, 0), W - 1);
            int cic = min(ci, Cin - 1);
            float v = X[(((img * Cin + cic) << logHW) + (iyc << logW)) + ixc];
            va[j] = ok ? v : 0.0f;
        }
#pragma unroll
        for (int j = 0; j < 8; ++j) {
            int i = tid + j * 256;
            lsA[i & 63][i >> 6] = (__bf16)va[j];
        }
        // ---- stage B: weights [co][ci*9+t], k-fast (contiguous) ----
        float vb[8];
#pragma unroll
        for (int j = 0; j < 8; ++j) {
            int i  = tid + j * 256;
            int kk = i & 31;
            int n  = i >> 5;
            int k  = k0 + kk;
            float v = Wt[(nBase + n) * Ktot + min(k, Ktot - 1)];
            vb[j] = (k < Ktot) ? v : 0.0f;
        }
#pragma unroll
        for (int j = 0; j < 8; ++j) {
            int i = tid + j * 256;
            lsBT[i >> 5][i & 31] = (__bf16)vb[j];
        }
        __syncthreads();

        // ---- fragments + 2x WMMA ----
        {
            int hiA = (lane >= 16) ? 8 : 0;
            int hiB = (lane >= 16) ? 8 : 0;
            v16bf af = frag_a_bf16(&lsA[wm + (lane & 15)][0], hiA);
            v16bf b0 = frag_b_bf16(&lsBT[wng + (lane & 15)][0], hiB);
            v16bf b1 = frag_b_bf16(&lsBT[wng + 16 + (lane & 15)][0], hiB);
            acc0 = __builtin_amdgcn_wmma_f32_16x16x32_bf16(false, af, false, b0,
                                                           (short)0, acc0, false, false);
            acc1 = __builtin_amdgcn_wmma_f32_16x16x32_bf16(false, af, false, b1,
                                                           (short)0, acc1, false, false);
        }
        __syncthreads();
    }

    // ---- epilogue: D element r -> row (r + 8*(lane>=16)), col = lane&15 ----
    const int rofs = (lane >= 16) ? 8 : 0;
#pragma unroll
    for (int half = 0; half < 2; ++half) {
        const v8f& acc = half ? acc1 : acc0;
        int ct   = wng + half * 16 + (lane & 15);   // channel within tile
        int ncol = nBase + ct;
        float bv = bias[ncol];
#pragma unroll
        for (int r = 0; r < 8; ++r) {
            int mr = mBase + wm + r + rofs;
            float y = (acc[r] + bv) * scale;
            if (doRelu) y = fmaxf(y, 0.0f);
            int img = mr >> logHW;
            int rem = mr & (HW - 1);
            int oy  = rem >> logW, ox = rem & (W - 1);
            if (POOL) {
                atomicAdd(&spool[ct][ox >> 3], y);
            } else {
                Y[(((img * Cout + ncol) << logHW) + (oy << logW)) + ox] = y;
            }
        }
    }

    if (POOL) {
        // block = 2 image rows (even oy) -> one pool row py; flush partials
        __syncthreads();
        int ch = tid >> 2, px = tid & 3;
        int c  = nBase + ch;
        int img = mBase >> logHW;
        int py  = ((mBase & (HW - 1)) >> logW) >> 3;
        atomicAdd(&pooled[((img * Cout + c) * 4 + py) * 4 + px],
                  spool[ch][px] * (1.0f / 64.0f));
    }
}

// ---------------------------------------------------------------------------
// Dense GEMM: Y[M,N] = A[M,K] @ Wt[N,K]^T + bias (+ReLU). Tiles are exact for
// all call sites, so tiles are staged with GLOBAL_LOAD_ASYNC_TO_LDS_B32
// (ASYNCcnt) straight into LDS as f32; converted to bf16 at fragment build.
// ---------------------------------------------------------------------------
__global__ __launch_bounds__(256)
void gemm_wmma(const float* __restrict__ A, const float* __restrict__ Wt,
               const float* __restrict__ bias, float* __restrict__ Y,
               int M, int K, int N, int doRelu)
{
    __shared__ __align__(16) float lsA [MT][LDAF];   // [m][k] f32
    __shared__ __align__(16) float lsBT[NTB][LDAF];  // [n][k] f32

    const int mBase = blockIdx.x * MT;
    const int nBase = blockIdx.y * NTB;
    const int tid   = threadIdx.x;
    const int lane  = tid & 31;
    const int wv    = tid >> 5;
    const int wm    = (wv & 3) * 16;
    const int wng   = (wv >> 2) * 32;

    v8f acc0 = {}, acc1 = {};

    for (int k0 = 0; k0 < K; k0 += KT) {
        // async-stage A tile (64x32, k-fast: coalesced)
#pragma unroll
        for (int j = 0; j < 8; ++j) {
            int i  = tid + j * 256;
            int kk = i & 31, r = i >> 5;
            const float* src = A + (size_t)(mBase + r) * K + (k0 + kk);
            unsigned dst = (unsigned)(uintptr_t)&lsA[r][kk];
            asm volatile("global_load_async_to_lds_b32 %0, %1, off"
                         :: "v"(dst), "v"((uint64_t)(uintptr_t)src) : "memory");
        }
        // async-stage B tile (64 rows x 32 k)
#pragma unroll
        for (int j = 0; j < 8; ++j) {
            int i  = tid + j * 256;
            int kk = i & 31, n = i >> 5;
            const float* src = Wt + (size_t)(nBase + n) * K + (k0 + kk);
            unsigned dst = (unsigned)(uintptr_t)&lsBT[n][kk];
            asm volatile("global_load_async_to_lds_b32 %0, %1, off"
                         :: "v"(dst), "v"((uint64_t)(uintptr_t)src) : "memory");
        }
        asm volatile("s_wait_asynccnt 0x0" ::: "memory");
        __syncthreads();

        {
            int hiA = (lane >= 16) ? 8 : 0;
        int hiB = (lane >= 16) ? 8 : 0;
            const float* ra = &lsA[wm + (lane & 15)][0];
            f32x4 a0 = *(const f32x4*)(ra + hiA);
            f32x4 a1 = *(const f32x4*)(ra + hiA + 4);
            f32x4 a2 = *(const f32x4*)(ra + hiA + 16);
            f32x4 a3 = *(const f32x4*)(ra + hiA + 20);
            v16bf af;
#pragma unroll
            for (int j = 0; j < 4; ++j) {
                af[j]      = (__bf16)a0[j];
                af[4 + j]  = (__bf16)a1[j];
                af[8 + j]  = (__bf16)a2[j];
                af[12 + j] = (__bf16)a3[j];
            }
            const float* rb0 = &lsBT[wng + (lane & 15)][0];
            const float* rb1 = &lsBT[wng + 16 + (lane & 15)][0];
            v16bf b0, b1;
#pragma unroll
            for (int j = 0; j < 4; ++j) {
                f32x4 t0 = *(const f32x4*)(rb0 + hiB + 4 * j + (j >> 1) * 8);
                f32x4 t1 = *(const f32x4*)(rb1 + hiB + 4 * j + (j >> 1) * 8);
#pragma unroll
                for (int q = 0; q < 4; ++q) {
                    b0[4 * j + q] = (__bf16)t0[q];
                    b1[4 * j + q] = (__bf16)t1[q];
                }
            }
            acc0 = __builtin_amdgcn_wmma_f32_16x16x32_bf16(false, af, false, b0,
                                                           (short)0, acc0, false, false);
            acc1 = __builtin_amdgcn_wmma_f32_16x16x32_bf16(false, af, false, b1,
                                                           (short)0, acc1, false, false);
        }
        __syncthreads();
    }

    const int rofs = (lane >= 16) ? 8 : 0;
#pragma unroll
    for (int half = 0; half < 2; ++half) {
        const v8f& acc = half ? acc1 : acc0;
        int ncol = nBase + wng + half * 16 + (lane & 15);
        float bv = bias[ncol];
#pragma unroll
        for (int r = 0; r < 8; ++r) {
            int mr = mBase + wm + r + rofs;
            float y = acc[r] + bv;
            if (doRelu) y = fmaxf(y, 0.0f);
            Y[(size_t)mr * N + ncol] = y;
        }
    }
}

// ---------------------------------------------------------------------------
// Small helper kernels
// ---------------------------------------------------------------------------
__global__ void maxpool2x2(const float* __restrict__ in, float* __restrict__ out,
                           int NCtot, int H, int W)
{
    int Ho = H >> 1, Wo = W >> 1;
    int idx = blockIdx.x * blockDim.x + threadIdx.x;
    int tot = NCtot * Ho * Wo;
    if (idx >= tot) return;
    int wo = idx % Wo;
    int t  = idx / Wo;
    int ho = t % Ho;
    int nc = t / Ho;
    const float* p = in + ((size_t)nc * H + ho * 2) * W + wo * 2;
    out[idx] = fmaxf(fmaxf(p[0], p[1]), fmaxf(p[W], p[W + 1]));
}

__global__ void avgpool_global(const float* __restrict__ in, float* __restrict__ out,
                               int NCtot, int HW)
{
    int idx = blockIdx.x * blockDim.x + threadIdx.x;
    if (idx >= NCtot) return;
    const float* p = in + (size_t)idx * HW;
    float s = 0.0f;
    for (int i = 0; i < HW; ++i) s += p[i];
    out[idx] = s / (float)HW;
}

__global__ void scorer_heads(const float* __restrict__ h, const float* __restrict__ shw,
                             const float* __restrict__ shb, const float* __restrict__ cw,
                             const float* __restrict__ cb,
                             float* __restrict__ scores, float* __restrict__ clsLog, int e)
{
    int idx = blockIdx.x * blockDim.x + threadIdx.x;
    if (idx >= 1024 * 11) return;
    int b = idx / 11, j = idx - b * 11;
    const float* hb = h + (size_t)b * 128;
    if (j == 10) {
        float s = shb[0];
        for (int k = 0; k < 128; ++k) s += hb[k] * shw[k];
        scores[b * 4 + e] = s;
    } else {
        float s = cb[j];
        const float* w = cw + (size_t)j * 128;
        for (int k = 0; k < 128; ++k) s += hb[k] * w[k];
        clsLog[(b * 4 + e) * 10 + j] = s;
    }
}

__global__ void compute_sn(const float* __restrict__ scores, const float* __restrict__ cls,
                           const int* __restrict__ targets, float* __restrict__ sn)
{
    int idx = blockIdx.x * blockDim.x + threadIdx.x;
    if (idx >= 1024 * 4) return;
    int b = idx >> 2;
    const float* l = cls + (size_t)idx * 10;
    float mx = l[0];
    for (int i = 1; i < 10; ++i) mx = fmaxf(mx, l[i]);
    float s = 0.0f;
    for (int i = 0; i < 10; ++i) s += expf(l[i] - mx);
    float gt = expf(l[targets[b]] - mx) / s;
    gt = fmaxf(gt, 1e-9f);
    sn[idx] = (scores[idx] + logf(gt)) * 10.0f;   // /T, T=0.1
}

// waterfall routing, exact reference semantics (one 1024-thread block)
__global__ __launch_bounds__(1024)
void route_waterfall(const float* __restrict__ sn, int* __restrict__ assign)
{
    __shared__ int wsum[32];
    __shared__ int capS[4];
    const int Ccap = 256;
    const int b    = threadIdx.x;
    const int lane = b & 31, wv = b >> 5;

    float snv[4];
#pragma unroll
    for (int e = 0; e < 4; ++e) snv[e] = sn[b * 4 + e];
    int asg[4] = {0, 0, 0, 0};
    int remaining = 1;
    if (b < 4) capS[b] = 0;
    __syncthreads();

    for (int it = 0; it < 15; ++it) {
        int quota = 1 << it;
        int best = 0; float bv = -INFINITY;
        for (int e = 0; e < 4; ++e) {
            float d  = fminf(fmaxf((float)capS[e] / (float)Ccap, 0.0f), 1.0f);
            float sv = (capS[e] >= Ccap) ? -INFINITY : (snv[e] - d);
            if (sv > bv) { bv = sv; best = e; }
        }
        for (int e = 0; e < 4; ++e) {
            int want = (remaining && best == e) ? 1 : 0;
            __syncthreads();
            unsigned mb = (unsigned)__ballot(want);
            if (lane == 0) wsum[wv] = __popc(mb);
            int lanePre = __popc(mb & ((1u << lane) - 1u));
            __syncthreads();
            int space = min(Ccap - capS[e], quota);
            int off = 0, tot = 0;
            for (int i = 0; i < 32; ++i) { int v = wsum[i]; tot += v; if (i < wv) off += v; }
            int rank = off + lanePre + want - 1;
            int sel  = want && (rank < space);
            if (sel) { asg[e] = 1; remaining = 0; }
            __syncthreads();
            if (b == 0) capS[e] += min(max(space, 0), tot);
        }
        __syncthreads();
    }
    int least = 0; float mv = (float)capS[0];
    for (int e = 1; e < 4; ++e) {
        float v = (float)capS[e];
        if (v < mv) { mv = v; least = e; }
    }
    if (remaining) asg[least] = 1;
#pragma unroll
    for (int e = 0; e < 4; ++e) assign[b * 4 + e] = asg[e];
}

__global__ void cls_combine(const float* __restrict__ z, const float* __restrict__ cw,
                            const float* __restrict__ cb, const int* __restrict__ assign,
                            float* __restrict__ out, int e)
{
    int idx = blockIdx.x * blockDim.x + threadIdx.x;
    if (idx >= 1024 * 10) return;
    int b = idx / 10, n = idx - b * 10;
    if (!assign[b * 4 + e]) return;
    const float* zb = z + (size_t)b * 256;
    const float* w  = cw + (size_t)n * 256;
    float s = cb[n];
    for (int k = 0; k < 256; ++k) s += zb[k] * w[k];
    out[idx] += s;
}

// ---------------------------------------------------------------------------
extern "C" void kernel_launch(void* const* d_in, const int* in_sizes, int n_in,
                              void* d_out, int out_size, void* d_ws, size_t ws_size,
                              hipStream_t stream)
{
    const float* x        = (const float*)d_in[0];
    const int*   targets  = (const int*)  d_in[1];
    const float* sc_w1    = (const float*)d_in[2];
    const float* sc_b1    = (const float*)d_in[3];
    const float* sc_w2    = (const float*)d_in[4];
    const float* sc_b2    = (const float*)d_in[5];
    const float* sh_w     = (const float*)d_in[6];
    const float* sh_b     = (const float*)d_in[7];
    const float* sc_cls_w = (const float*)d_in[8];
    const float* sc_cls_b = (const float*)d_in[9];
    const float* e_w1     = (const float*)d_in[10];
    const float* e_b1     = (const float*)d_in[11];
    const float* e_w2     = (const float*)d_in[12];
    const float* e_b2     = (const float*)d_in[13];
    const float* e_w3     = (const float*)d_in[14];
    const float* e_b3     = (const float*)d_in[15];
    const float* e_w4     = (const float*)d_in[16];
    const float* e_b4     = (const float*)d_in[17];
    const float* proj_w   = (const float*)d_in[18];
    const float* proj_b   = (const float*)d_in[19];
    const float* cls_w    = (const float*)d_in[20];
    const float* cls_b    = (const float*)d_in[21];
    float* out = (float*)d_out;

    const int B = 1024, E = 4, NC = 10, SH = 128;
    const int CHUNK = 128, NCHUNK = B / CHUNK;
    const float BNv = 1.0f / sqrtf(1.0f + 1e-5f);

    float* ws = (float*)d_ws;
    size_t o = 0;
    float* pooled  = ws + o; o += (size_t)B * SH * 16;
    float* hfc     = ws + o; o += (size_t)B * SH;
    float* scoresA = ws + o; o += (size_t)B * E;
    float* clsLog  = ws + o; o += (size_t)B * E * NC;
    float* snb     = ws + o; o += (size_t)B * E;
    int*   assign  = (int*)(ws + o); o += (size_t)B * E;
    float* z0      = ws + o; o += (size_t)B * 256;
    float* zproj   = ws + o; o += (size_t)B * 256;
    float* buf1    = ws + o; o += (size_t)CHUNK * 64 * 32 * 32;
    float* buf2    = ws + o;

    hipMemsetAsync(out, 0, (size_t)B * NC * sizeof(float), stream);

    // ---- scorer trunk + heads ----
    for (int e = 0; e < E; ++e) {
        hipMemsetAsync(pooled, 0, (size_t)B * SH * 16 * sizeof(float), stream);
        dim3 gc((B * 32 * 32) / MT, SH / NTB);
        conv3x3_wmma<1><<<gc, 256, 0, stream>>>(x, sc_w1 + (size_t)e * SH * 27,
            sc_b1 + e * SH, nullptr, pooled, B, 3, SH, 5, 1.0f, 1);
        dim3 gf(B / MT, SH / NTB);
        gemm_wmma<<<gf, 256, 0, stream>>>(pooled, sc_w2 + (size_t)e * SH * 2048,
            sc_b2 + e * SH, hfc, B, SH * 16, SH, 1);
        scorer_heads<<<(B * 11 + 255) / 256, 256, 0, stream>>>(hfc,
            sh_w + (size_t)e * SH, sh_b + e, sc_cls_w + (size_t)e * NC * SH,
            sc_cls_b + e * NC, scoresA, clsLog, e);
    }
    compute_sn<<<(B * E + 255) / 256, 256, 0, stream>>>(scoresA, clsLog, targets, snb);
    route_waterfall<<<1, 1024, 0, stream>>>(snb, assign);

    // ---- expert encoders ----
    for (int e = 0; e < E; ++e) {
        for (int c = 0; c < NCHUNK; ++c) {
            const float* xc = x + (size_t)c * CHUNK * 3 * 1024;
            dim3 g1((CHUNK * 1024) / MT, 64 / NTB);
            conv3x3_wmma<0><<<g1, 256, 0, stream>>>(xc, e_w1 + (size_t)e * 64 * 27,
                e_b1 + e * 64, buf1, nullptr, CHUNK, 3, 64, 5, BNv, 1);
            conv3x3_wmma<0><<<g1, 256, 0, stream>>>(buf1, e_w2 + (size_t)e * 64 * 576,
                e_b2 + e * 64, buf2, nullptr, CHUNK, 64, 64, 5, BNv, 1);
            maxpool2x2<<<(CHUNK * 64 * 16 * 16 + 255) / 256, 256, 0, stream>>>(
                buf2, buf1, CHUNK * 64, 32, 32);
            dim3 g3((CHUNK * 256) / MT, 128 / NTB);
            conv3x3_wmma<0><<<g3, 256, 0, stream>>>(buf1, e_w3 + (size_t)e * 128 * 576,
                e_b3 + e * 128, buf2, nullptr, CHUNK, 64, 128, 4, BNv, 1);
            maxpool2x2<<<(CHUNK * 128 * 8 * 8 + 255) / 256, 256, 0, stream>>>(
                buf2, buf1, CHUNK * 128, 16, 16);
            dim3 g4((CHUNK * 64) / MT, 256 / NTB);
            conv3x3_wmma<0><<<g4, 256, 0, stream>>>(buf1, e_w4 + (size_t)e * 256 * 1152,
                e_b4 + e * 256, buf2, nullptr, CHUNK, 128, 256, 3, BNv, 1);
            avgpool_global<<<(CHUNK * 256 + 255) / 256, 256, 0, stream>>>(
                buf2, z0 + (size_t)c * CHUNK * 256, CHUNK * 256, 64);
        }
        dim3 gp(B / MT, 256 / NTB);
        gemm_wmma<<<gp, 256, 0, stream>>>(z0, proj_w + (size_t)e * 256 * 256,
            proj_b + e * 256, zproj, B, 256, 256, 0);
        cls_combine<<<(B * NC + 255) / 256, 256, 0, stream>>>(zproj,
            cls_w + (size_t)e * NC * 256, cls_b + e * NC, assign, out, e);
    }

    (void)in_sizes; (void)n_in; (void)out_size; (void)ws_size;
}
